// VisionMambaBlock_14156212207681
// MI455X (gfx1250) — matrix operand order repaired and checked by reading, not compiled, not run on hardware
//
#include <hip/hip_runtime.h>

typedef __bf16 bf16_t;
typedef __attribute__((ext_vector_type(16))) __bf16 bf16x16;
typedef __attribute__((ext_vector_type(8)))  __bf16 bf16x8;
typedef __attribute__((ext_vector_type(4)))  __bf16 bf16x4;
typedef __attribute__((ext_vector_type(8)))  float  f32x8;

#define DEV static __device__ __forceinline__

// ---- model dims (compile-time) ----
constexpr int D_MODEL = 512;
constexpr int D_INNER = 1024;
constexpr int DT_RANK = 32;
constexpr int BB      = 4;
constexpr int NSEQ    = 576;
constexpr int MTOK    = BB * NSEQ;   // 2304 tokens

DEV bf16_t f2bf(float f) {
  unsigned u = __builtin_bit_cast(unsigned, f);
  unsigned r = (u + 0x7FFFu + ((u >> 16) & 1u)) >> 16;
  unsigned short s = (unsigned short)r;
  return __builtin_bit_cast(bf16_t, s);
}
DEV float silu_f(float x) { return x / (1.f + __expf(-x)); }
DEV float gelu_tanh(float x) {
  float x3 = x * x * x;
  return 0.5f * x * (1.f + tanhf(0.7978845608028654f * (x + 0.044715f * x3)));
}

// ---- CDNA5 async global->LDS copy (16B per lane), tracked by ASYNCcnt ----
DEV void async_b128_to_lds(bf16_t* lds_dst, const bf16_t* gsrc) {
  unsigned loff = (unsigned)(uintptr_t)lds_dst;      // low 32b of generic = LDS offset
  unsigned long long ga = (unsigned long long)(uintptr_t)gsrc;
  asm volatile("global_load_async_to_lds_b128 %0, %1, off"
               :: "v"(loff), "v"(ga) : "memory");
}
DEV void wait_async0() { asm volatile("s_wait_asynccnt 0x0" ::: "memory"); }

// =====================================================================
// WMMA GEMM: C[M,N] = act(A_bf16[M,K] @ W_f32[N,K]^T + bias) (+ res)
// block tile 128x128, 8 waves (4 along M x 2 along N), wave tile 32x64
// A tile staged via GLOBAL_LOAD_ASYNC_TO_LDS_B128; B (f32 weights)
// converted to bf16 in flight on the synchronous path.
// =====================================================================
constexpr int BM = 128, BN = 128, BK = 32;
constexpr int LDS_A = 40, LDS_B = 40;   // bf16 elems; 80B row stride (16B aligned)

__global__ __launch_bounds__(256)
void gemm_bf16_wmma(const bf16_t* __restrict__ A, int lda,
                    const float* __restrict__ W,          // (Nout, K) row-major f32
                    float* __restrict__ C, int ldc,       // optional f32 out
                    bf16_t* __restrict__ Cbf, int ldcbf,  // optional bf16 mirror
                    const float* __restrict__ bias,       // optional
                    const float* __restrict__ res, int ldres, // optional residual add
                    int M, int Nout, int K, int act)
{
  __shared__ __align__(32) bf16_t sA[BM * LDS_A];
  __shared__ __align__(32) bf16_t sB[BN * LDS_B];

  const int tid  = threadIdx.x;
  const int lane = tid & 31;
  const int wave = tid >> 5;
  const int wm   = wave & 3;     // 4 waves along M -> 32 rows each
  const int wn   = wave >> 2;    // 2 waves along N -> 64 cols each
  const int m0   = blockIdx.x * BM;
  const int n0   = blockIdx.y * BN;
  const int lh   = lane >> 4;    // lane half (0/1)
  const int l16  = lane & 15;

  // fixed staging coordinates (2 A-chunks, 4 B-chunks per thread)
  const int ar0 = tid >> 2,        ak0 = (tid & 3) * 8;          // A chunk 0
  const int ar1 = (tid + 256) >> 2, ak1 = ((tid + 256) & 3) * 8; // A chunk 1

  f32x8 acc[2][4];
  #pragma unroll
  for (int i = 0; i < 2; ++i)
    #pragma unroll
    for (int j = 0; j < 4; ++j)
      #pragma unroll
      for (int e = 0; e < 8; ++e) acc[i][j][e] = 0.f;

  for (int k0 = 0; k0 < K; k0 += BK) {
    // ---- stage A tile (BM x BK) bf16 via async DMA into LDS ----
    async_b128_to_lds(sA + ar0 * LDS_A + ak0,
                      A + (size_t)(m0 + ar0) * lda + k0 + ak0);
    async_b128_to_lds(sA + ar1 * LDS_A + ak1,
                      A + (size_t)(m0 + ar1) * lda + k0 + ak1);

    // ---- stage B tile (BN x BK): f32 weights -> bf16 (synchronous) ----
    #pragma unroll
    for (int q = 0; q < 4; ++q) {
      int c  = tid + q * 256;
      int r  = c >> 3;
      int kk = (c & 7) * 4;
      float w0 = 0.f, w1 = 0.f, w2 = 0.f, w3 = 0.f;
      if (n0 + r < Nout) {
        const float4 w = *reinterpret_cast<const float4*>(W + (size_t)(n0 + r) * K + k0 + kk);
        w0 = w.x; w1 = w.y; w2 = w.z; w3 = w.w;
      }
      bf16x4 o; o[0] = f2bf(w0); o[1] = f2bf(w1); o[2] = f2bf(w2); o[3] = f2bf(w3);
      *reinterpret_cast<bf16x4*>(sB + r * LDS_B + kk) = o;
    }

    // prefetch next weight K-tile into cache (global_prefetch_b8)
    if (k0 + BK < K) {
      int pr = n0 + (tid >> 1);
      if (pr < Nout) __builtin_prefetch(W + (size_t)pr * K + k0 + BK, 0, 1);
    }

    wait_async0();       // our async LDS writes landed
    __syncthreads();     // everyone's tile visible

    // ---- per-lane fragments (CDNA5 16-bit WMMA layouts) ----
    // A 16x32: elems 0..7 -> K = lh*8 + j ; elems 8..15 -> K = 16 + lh*8 + j
    bf16x16 af[2];
    #pragma unroll
    for (int i = 0; i < 2; ++i) {
      const bf16_t* p = sA + (wm * 32 + i * 16 + l16) * LDS_A;
      bf16x8 lo = *reinterpret_cast<const bf16x8*>(p + lh * 8);
      bf16x8 hi = *reinterpret_cast<const bf16x8*>(p + 16 + lh * 8);
      #pragma unroll
      for (int e = 0; e < 8; ++e) { af[i][e] = lo[e]; af[i][8 + e] = hi[e]; }
    }
    // B 32x16: elems 0..7 -> K = lh*8 + j ; elems 8..15 -> K = 16 + lh*8 + j, col = l16
    bf16x16 bfr[4];
    #pragma unroll
    for (int j = 0; j < 4; ++j) {
      const bf16_t* p = sB + (wn * 64 + j * 16 + l16) * LDS_B;
      bf16x8 lo = *reinterpret_cast<const bf16x8*>(p + lh * 8);
      bf16x8 hi = *reinterpret_cast<const bf16x8*>(p + 16 + lh * 8);
      #pragma unroll
      for (int e = 0; e < 8; ++e) { bfr[j][e] = lo[e]; bfr[j][8 + e] = hi[e]; }
    }
    #pragma unroll
    for (int i = 0; i < 2; ++i)
      #pragma unroll
      for (int j = 0; j < 4; ++j)
        acc[i][j] = __builtin_amdgcn_wmma_f32_16x16x32_bf16(
            false, af[i], false, bfr[j], (short)0, acc[i][j], false, false);
    __syncthreads();
  }

  // ---- epilogue: C/D layout -> row = r + 8*lh, col = l16 ----
  const int rl = lh * 8;
  #pragma unroll
  for (int i = 0; i < 2; ++i) {
    #pragma unroll
    for (int j = 0; j < 4; ++j) {
      int colg = n0 + wn * 64 + j * 16 + l16;
      if (colg >= Nout) continue;
      float bv = bias ? bias[colg] : 0.f;
      #pragma unroll
      for (int r = 0; r < 8; ++r) {
        int rowg = m0 + wm * 32 + i * 16 + rl + r;
        if (rowg >= M) continue;
        float v = acc[i][j][r] + bv;
        if (act == 1) v = gelu_tanh(v);
        if (res) v += res[(size_t)rowg * ldres + colg];
        if (C)   C[(size_t)rowg * ldc + colg] = v;
        if (Cbf) Cbf[(size_t)rowg * ldcbf + colg] = f2bf(v);
      }
    }
  }
}

// =====================================================================
// LayerNorm over D_MODEL=512, one token per 256-thread block, bf16 out
// =====================================================================
__global__ __launch_bounds__(256)
void layernorm_bf16_kernel(const float* __restrict__ x, const float* __restrict__ g,
                           const float* __restrict__ b, bf16_t* __restrict__ out)
{
  __shared__ float red[256];
  __shared__ float s_mean, s_rstd;
  const int row = blockIdx.x;
  const int tid = threadIdx.x;
  const float* xr = x + (size_t)row * D_MODEL;
  float v0 = xr[tid], v1 = xr[tid + 256];
  red[tid] = v0 + v1;
  __syncthreads();
  for (int s = 128; s > 0; s >>= 1) { if (tid < s) red[tid] += red[tid + s]; __syncthreads(); }
  if (tid == 0) s_mean = red[0] * (1.f / D_MODEL);
  __syncthreads();
  float m = s_mean;
  float d0 = v0 - m, d1 = v1 - m;
  red[tid] = d0 * d0 + d1 * d1;
  __syncthreads();
  for (int s = 128; s > 0; s >>= 1) { if (tid < s) red[tid] += red[tid + s]; __syncthreads(); }
  if (tid == 0) s_rstd = rsqrtf(red[0] * (1.f / D_MODEL) + 1e-5f);
  __syncthreads();
  float rs = s_rstd;
  bf16_t* orow = out + (size_t)row * D_MODEL;
  orow[tid]       = f2bf(d0 * rs * g[tid]       + b[tid]);
  orow[tid + 256] = f2bf(d1 * rs * g[tid + 256] + b[tid + 256]);
}

// =====================================================================
// Depthwise causal conv1d (k=4) + bias + SiLU on u = xz[:, :D_INNER].
// rev=1: operate in flipped-sequence domain.
// =====================================================================
__global__ __launch_bounds__(256)
void conv_silu_kernel(const float* __restrict__ xz, const float* __restrict__ cw,
                      const float* __restrict__ cb, float* __restrict__ uf,
                      bf16_t* __restrict__ ubf, int rev)
{
  int idx = blockIdx.x * 256 + threadIdx.x;
  if (idx >= MTOK * D_INNER) return;
  int c   = idx & (D_INNER - 1);
  int tok = idx >> 10;
  int b = tok / NSEQ, n = tok % NSEQ;
  float acc = cb[c];
  #pragma unroll
  for (int j = 0; j < 4; ++j) {
    int t = n - 3 + j;                 // causal tap in (possibly flipped) time
    if (t >= 0) {
      int src = rev ? (NSEQ - 1 - t) : t;
      acc += cw[c * 4 + j] * xz[(size_t)(b * NSEQ + src) * (2 * D_INNER) + c];
    }
  }
  float s = silu_f(acc);
  uf[idx]  = s;
  ubf[idx] = f2bf(s);
}

// =====================================================================
// Selective scan: per (b, channel d) sequential over N; 16 f32 states in
// registers; per-step B/C (32 floats) broadcast through LDS.
// =====================================================================
__global__ __launch_bounds__(256)
void scan_kernel(const float* __restrict__ dtp, const float* __restrict__ uf,
                 const float* __restrict__ xdbl, const float* __restrict__ Alog,
                 const float* __restrict__ Dp, float* __restrict__ ys)
{
  const int d = blockIdx.x * 256 + threadIdx.x;   // channel 0..1023
  const int b = blockIdx.y;
  __shared__ float sBC[32];
  float Areal[16];
  #pragma unroll
  for (int s = 0; s < 16; ++s) Areal[s] = -__expf(Alog[d * 16 + s]);
  const float Dd = Dp[d];
  float h[16];
  #pragma unroll
  for (int s = 0; s < 16; ++s) h[s] = 0.f;

  for (int n = 0; n < NSEQ; ++n) {
    size_t base = (size_t)(b * NSEQ + n);
    if (threadIdx.x < 32) sBC[threadIdx.x] = xdbl[base * 64 + 32 + threadIdx.x];
    __syncthreads();
    float dtv = dtp[base * D_INNER + d];
    dtv = (dtv > 20.f) ? dtv : __logf(1.f + __expf(dtv));   // softplus
    float uv = uf[base * D_INNER + d];
    float y = 0.f;
    #pragma unroll
    for (int s = 0; s < 16; ++s) {
      float dA  = __expf(dtv * Areal[s]);
      float dBu = dtv * sBC[s] * uv;
      h[s] = dA * h[s] + dBu;
      y += h[s] * sBC[16 + s];
    }
    ys[base * D_INNER + d] = y + uv * Dd;
    __syncthreads();
  }
}

// =====================================================================
// Gate: out = ys * silu(z); rev=1 also performs the flip-back so the
// output is in natural token order for both directions.
// =====================================================================
__global__ __launch_bounds__(256)
void gate_kernel(const float* __restrict__ ys, const float* __restrict__ xz,
                 bf16_t* __restrict__ out, int rev)
{
  int idx = blockIdx.x * 256 + threadIdx.x;
  if (idx >= MTOK * D_INNER) return;
  int c   = idx & (D_INNER - 1);
  int tok = idx >> 10;
  int b = tok / NSEQ, n = tok % NSEQ;
  int ysrow = rev ? (b * NSEQ + (NSEQ - 1 - n)) : tok;
  float z = xz[(size_t)tok * (2 * D_INNER) + D_INNER + c];
  float v = ys[(size_t)ysrow * D_INNER + c] * silu_f(z);
  out[(size_t)tok * D_INNER + c] = f2bf(v);
}

// =====================================================================
extern "C" void kernel_launch(void* const* d_in, const int* in_sizes, int n_in,
                              void* d_out, int out_size, void* d_ws, size_t ws_size,
                              hipStream_t stream)
{
  (void)in_sizes; (void)n_in; (void)out_size; (void)ws_size;
  const float* x       = (const float*)d_in[0];
  const float* g1      = (const float*)d_in[1];
  const float* b1      = (const float*)d_in[2];
  const float* g2      = (const float*)d_in[3];
  const float* b2      = (const float*)d_in[4];
  const float* merge_W = (const float*)d_in[5];
  const float* merge_b = (const float*)d_in[6];
  const float* ffn_W1  = (const float*)d_in[7];
  const float* ffn_b1  = (const float*)d_in[8];
  const float* ffn_W2  = (const float*)d_in[9];
  const float* ffn_b2  = (const float*)d_in[10];
  const float* Win[2]   = {(const float*)d_in[11], (const float*)d_in[20]};
  const float* convw[2] = {(const float*)d_in[12], (const float*)d_in[21]};
  const float* convb[2] = {(const float*)d_in[13], (const float*)d_in[22]};
  const float* Wx[2]    = {(const float*)d_in[14], (const float*)d_in[23]};
  const float* Wdt[2]   = {(const float*)d_in[15], (const float*)d_in[24]};
  const float* bdt[2]   = {(const float*)d_in[16], (const float*)d_in[25]};
  const float* Alog[2]  = {(const float*)d_in[17], (const float*)d_in[26]};
  const float* Dp[2]    = {(const float*)d_in[18], (const float*)d_in[27]};
  const float* Wout[2]  = {(const float*)d_in[19], (const float*)d_in[28]};

  // ---- workspace carve-up (deterministic) ----
  char* ws = (char*)d_ws;
  size_t off = 0;
  auto alloc = [&](size_t bytes) -> void* {
    void* p = ws + off;
    off += (bytes + 255) & ~(size_t)255;
    return p;
  };
  bf16_t* xn_bf = (bf16_t*)alloc((size_t)MTOK * D_MODEL * 2);
  float*  xz[2]; float* uf[2]; bf16_t* ubf[2]; float* xdbl[2]; bf16_t* xdblbf[2];
  float*  dtp[2]; float* ys[2]; bf16_t* yg[2];
  for (int d = 0; d < 2; ++d) {
    xz[d]     = (float*) alloc((size_t)MTOK * 2 * D_INNER * 4);
    uf[d]     = (float*) alloc((size_t)MTOK * D_INNER * 4);
    ubf[d]    = (bf16_t*)alloc((size_t)MTOK * D_INNER * 2);
    xdbl[d]   = (float*) alloc((size_t)MTOK * 64 * 4);
    xdblbf[d] = (bf16_t*)alloc((size_t)MTOK * 64 * 2);
    dtp[d]    = (float*) alloc((size_t)MTOK * D_INNER * 4);
    ys[d]     = (float*) alloc((size_t)MTOK * D_INNER * 4);
    yg[d]     = (bf16_t*)alloc((size_t)MTOK * D_INNER * 2);
  }
  bf16_t* xmcat_bf = (bf16_t*)alloc((size_t)MTOK * 2 * D_MODEL * 2);
  float*  out1     = (float*) alloc((size_t)MTOK * D_MODEL * 4);
  bf16_t* xn2_bf   = (bf16_t*)alloc((size_t)MTOK * D_MODEL * 2);
  bf16_t* h_bf     = (bf16_t*)alloc((size_t)MTOK * 4 * D_MODEL * 2);

  dim3 blk(256);
  auto gemm = [&](const bf16_t* A, int lda, const float* W, float* C, int ldc,
                  bf16_t* Cbf, int ldcbf, const float* bias,
                  const float* res, int ldres, int M_, int N_, int K_, int act) {
    dim3 grid((M_ + BM - 1) / BM, (N_ + BN - 1) / BN);
    gemm_bf16_wmma<<<grid, blk, 0, stream>>>(A, lda, W, C, ldc, Cbf, ldcbf,
                                             bias, res, ldres, M_, N_, K_, act);
  };
  const int gridEW = (MTOK * D_INNER + 255) / 256;

  // LN1 -> bf16 activations
  layernorm_bf16_kernel<<<MTOK, blk, 0, stream>>>(x, g1, b1, xn_bf);

  for (int d = 0; d < 2; ++d) {
    // xz = xn @ Win^T   (M x 2048)
    gemm(xn_bf, D_MODEL, Win[d], xz[d], 2 * D_INNER, nullptr, 0,
         nullptr, nullptr, 0, MTOK, 2 * D_INNER, D_MODEL, 0);
    // u = silu(causal_conv(xz[:, :1024]) + cb)  (flipped domain for d==1)
    conv_silu_kernel<<<gridEW, blk, 0, stream>>>(xz[d], convw[d], convb[d],
                                                 uf[d], ubf[d], d);
    // x_dbl = u @ Wx^T   (M x 64): dt | B | C
    gemm(ubf[d], D_INNER, Wx[d], xdbl[d], 64, xdblbf[d], 64,
         nullptr, nullptr, 0, MTOK, 64, D_INNER, 0);
    // dtproj = dt @ Wdt^T + bdt   (M x 1024); softplus applied in scan
    gemm(xdblbf[d], 64, Wdt[d], dtp[d], D_INNER, nullptr, 0,
         bdt[d], nullptr, 0, MTOK, D_INNER, DT_RANK, 0);
    // selective scan
    scan_kernel<<<dim3(D_INNER / 256, BB), blk, 0, stream>>>(
        dtp[d], uf[d], xdbl[d], Alog[d], Dp[d], ys[d]);
    // y * silu(z), flip-back for reverse direction
    gate_kernel<<<gridEW, blk, 0, stream>>>(ys[d], xz[d], yg[d], d);
    // xf/xb = y @ Wout^T  -> concat buffer (M x 1024), bf16
    gemm(yg[d], D_INNER, Wout[d], nullptr, 0, xmcat_bf + d * D_MODEL, 2 * D_MODEL,
         nullptr, nullptr, 0, MTOK, D_MODEL, D_INNER, 0);
  }

  // out1 = x + (concat @ merge_W^T + merge_b)
  gemm(xmcat_bf, 2 * D_MODEL, merge_W, out1, D_MODEL, nullptr, 0,
       merge_b, x, D_MODEL, MTOK, D_MODEL, 2 * D_MODEL, 0);
  // LN2
  layernorm_bf16_kernel<<<MTOK, blk, 0, stream>>>(out1, g2, b2, xn2_bf);
  // h = gelu(xn2 @ W1^T + b1)  (bf16 mirror only)
  gemm(xn2_bf, D_MODEL, ffn_W1, nullptr, 0, h_bf, 4 * D_MODEL,
       ffn_b1, nullptr, 0, MTOK, 4 * D_MODEL, D_MODEL, 1);
  // out = out1 + (h @ W2^T + b2)
  gemm(h_bf, 4 * D_MODEL, ffn_W2, (float*)d_out, D_MODEL, nullptr, 0,
       ffn_b2, out1, D_MODEL, MTOK, D_MODEL, 4 * D_MODEL, 0);
}
